// TorchMHADiffusionLM_75883482186047
// MI455X (gfx1250) — matrix operand order repaired and checked by reading, not compiled, not run on hardware
//
#include <hip/hip_runtime.h>

// ---------------------------------------------------------------------------
// MI455X / gfx1250 fused MHA (RoPE + learned-q-transform + flash attention +
// output projection). All matmuls on v_wmma_f32_16x16x32_f16 (wave32).
// Intermediates stored in WMMA fragment-major layout (2 x ds_load_b128 per
// fragment); GEMM loops software-pipelined (next tile's global_load_b128s
// overlap current WMMAs); Q tile staged via the Tensor Data Mover.
// ---------------------------------------------------------------------------

typedef __attribute__((ext_vector_type(16))) _Float16 v16h;
typedef __attribute__((ext_vector_type(8)))  float    v8f;

union FragF { v8f v; float e[8]; };
union Pack4 { _Float16 h[4]; uint2 u; };

#define NB 2
#define NH 16
#define NS 2048
#define ND 64
#define DM 1024   // NH*ND

__device__ __forceinline__ v8f wmma16(v16h a, v16h b, v8f c) {
    return __builtin_amdgcn_wmma_f32_16x16x32_f16(
        false, a, false, b, (short)0, c, false, false);
}

// Fragment-major index within one 512-half fragment (ISA 7.12.2 inverse maps).
__device__ __forceinline__ int fragA_idx(int m, int k) {
    return ((m & 15) + (((k >> 3) & 1) << 4)) * 16 + (((k >> 4) & 1) << 3) + (k & 7);
}
__device__ __forceinline__ int fragB_idx(int k, int n) {
    return ((n & 15) + (((k >> 3) & 1) << 4)) * 16 + (((k >> 4) & 1) << 3) + (k & 7);
}
__device__ __forceinline__ v16h frag_ld(const _Float16* f, int lane) {
    return *(const v16h*)(f + lane * 16);   // 2 x ds_load_b128
}

// ---------------------------------------------------------------------------
// Tensor Data Mover: contiguous (nelems x 2B) global -> LDS copy, one wave.
// D# per CDNA5 ISA (2-D descriptor, 1 line).
// ---------------------------------------------------------------------------
#if defined(__has_builtin)
#if __has_builtin(__builtin_amdgcn_tensor_load_to_lds)
#define HAVE_TDM 1
#endif
#endif

#ifdef HAVE_TDM
typedef unsigned int tdm_v4u __attribute__((ext_vector_type(4)));
typedef int          tdm_v4i __attribute__((ext_vector_type(4)));
typedef int          tdm_v8i __attribute__((ext_vector_type(8)));

__device__ __forceinline__ void tdm_copy_to_lds(void* lds, const void* gmem,
                                                unsigned nelems /* 2B units */) {
    unsigned           lds_off = (unsigned)(uintptr_t)lds;   // low 32b = LDS offset
    unsigned long long ga      = (unsigned long long)(uintptr_t)gmem;
    tdm_v4u g0 = { 1u,                                        // count=1, user D#
                   lds_off,                                   // lds_addr [63:32]
                   (unsigned)(ga & 0xFFFFFFFFu),              // global_addr lo
                   (unsigned)((ga >> 32) & 0x1FFFFFFu) | (2u << 30) };  // hi | type=2
    tdm_v8i g1 = { (int)(1u << 16),                           // data_size=2B
                   (int)((nelems & 0xFFFFu) << 16),           // tensor_dim0[15:0]
                   (int)((nelems >> 16) | (1u << 16)),        // dim0[31:16] | dim1=1
                   (int)((nelems & 0xFFFFu) << 16),           // tile_dim0
                   1,                                         // tile_dim1=1
                   (int)nelems,                               // dim0_stride lo
                   0, 0 };
    tdm_v4i z4 = { 0, 0, 0, 0 };
#if __clang_major__ >= 23
    tdm_v8i z8 = { 0, 0, 0, 0, 0, 0, 0, 0 };
    __builtin_amdgcn_tensor_load_to_lds(g0, g1, z4, z4, z8, 0);
#else
    __builtin_amdgcn_tensor_load_to_lds(g0, g1, z4, z4, 0);
#endif
}
#endif

// ---------------------------------------------------------------------------
// Kernel 1: RoPE(q), RoPE(k), v->f16; outputs pre-swizzled fragment-major.
// ---------------------------------------------------------------------------
__global__ void k_rope(const float* __restrict__ q, const float* __restrict__ k,
                       const float* __restrict__ v,
                       _Float16* __restrict__ qr_t, _Float16* __restrict__ kr_t,
                       _Float16* __restrict__ vh_t) {
    int i = blockIdx.x * blockDim.x + threadIdx.x;
    if (i >= NB * NH * NS * ND) return;
    int d   = i & 63;
    int pos = (i >> 6) & (NS - 1);
    int bh  = i >> 17;
    int jj  = d & 31;
    float inv = __powf(10000.0f, -(float)(2 * jj) * (1.0f / 64.0f));
    float ang = (float)pos * inv;
    float c = cosf(ang), sn = sinf(ang);
    float qv = q[i], kv = k[i];
    float qrot = (d < 32) ? -q[i + 32] : q[i - 32];
    float krot = (d < 32) ? -k[i + 32] : k[i - 32];
    _Float16 qh  = (_Float16)(qv * c + qrot * sn);
    _Float16 kh  = (_Float16)(kv * c + krot * sn);
    _Float16 vhv = (_Float16)v[i];
    size_t base = (size_t)bh * NS * ND;
    qr_t[base + (size_t)(pos >> 5) * 2048
              + (d >> 4) * 512 + fragB_idx(pos & 31, d & 15)] = qh;
    kr_t[base + (size_t)(pos >> 6) * 4096
              + ((d >> 5) * 4 + ((pos >> 4) & 3)) * 512
              + fragB_idx(d & 31, pos & 15)] = kh;
    vh_t[base + (size_t)(pos >> 6) * 4096
              + (((pos >> 5) & 1) * 4 + (d >> 4)) * 512
              + fragB_idx(pos & 31, d & 15)] = vhv;
}

// ---------------------------------------------------------------------------
// Kernel 2: out_w -> f16, pre-swizzled B-layout tiles (K=in-feat, N=out-feat)
// ---------------------------------------------------------------------------
__global__ void k_cvtw(const float* __restrict__ w, _Float16* __restrict__ wh_t, int n) {
    int i = blockIdx.x * blockDim.x + threadIdx.x;
    if (i >= n) return;
    int nf = i >> 10;
    int kf = i & 1023;
    wh_t[((size_t)(kf >> 5) * (DM >> 6) + (nf >> 6)) * 2048
         + ((nf >> 4) & 3) * 512 + fragB_idx(kf & 31, nf & 15)] = (_Float16)w[i];
}

// ---------------------------------------------------------------------------
// Kernel 3: qt = (W @ qr) * 1/sqrt(d).  M=2048,N=64,K=2048 per (b,h).
// ---------------------------------------------------------------------------
__global__ void k_qt(const float* __restrict__ mask, const _Float16* __restrict__ qr_t,
                     _Float16* __restrict__ qt_t) {
    __shared__ __align__(32) _Float16 At[2048];   // 4 A-frags (64x32)
    __shared__ __align__(32) _Float16 Bt[2048];   // 4 B-frags (32x64)
    int m0 = blockIdx.x * 64, bh = blockIdx.y, hh = bh & (NH - 1);
    int tid = threadIdx.x, lane = tid & 31, w = tid >> 5;
    const float*    Wp = mask + (size_t)hh * NS * NS;
    const _Float16* Qp = qr_t + (size_t)bh * NS * ND;

    FragF acc[4];
#pragma unroll
    for (int c = 0; c < 4; ++c)
#pragma unroll
        for (int e = 0; e < 8; ++e) acc[c].e[e] = 0.0f;

    float4 aw[4];
    uint4  bw[2];
    auto load_tiles = [&](int k0) {
#pragma unroll
        for (int u = 0; u < 4; ++u) {            // exactly 512/128 groups
            int g = tid + (u << 7);
            int r = g >> 3, c4 = (g & 7) << 2;
            aw[u] = *(const float4*)(Wp + (size_t)(m0 + r) * NS + k0 + c4);
        }
        const uint4* s4 = (const uint4*)(Qp + (size_t)(k0 >> 5) * 2048);
        bw[0] = s4[tid];
        bw[1] = s4[tid + 128];
    };
    auto store_tiles = [&]() {
#pragma unroll
        for (int u = 0; u < 4; ++u) {
            int g = tid + (u << 7);
            int r = g >> 3, c4 = (g & 7) << 2;
            Pack4 p;
            p.h[0] = (_Float16)aw[u].x; p.h[1] = (_Float16)aw[u].y;
            p.h[2] = (_Float16)aw[u].z; p.h[3] = (_Float16)aw[u].w;
            *(uint2*)(At + (r >> 4) * 512 + fragA_idx(r & 15, c4)) = p.u;
        }
        uint4* d4 = (uint4*)Bt;
        d4[tid] = bw[0];
        d4[tid + 128] = bw[1];
    };

    load_tiles(0);
    for (int k0 = 0; k0 < NS; k0 += 32) {
        __syncthreads();
        store_tiles();
        __syncthreads();
        if (k0 + 32 < NS) {
            if (k0 + 64 < NS && tid < 64)        // global_prefetch_b8, 2 tiles ahead
                __builtin_prefetch(Wp + (size_t)(m0 + tid) * NS + k0 + 64, 0, 1);
            load_tiles(k0 + 32);                 // overlaps with WMMAs below
        }
        v16h a = frag_ld(At + w * 512, lane);
        v16h bf[4];
#pragma unroll
        for (int c = 0; c < 4; ++c) bf[c] = frag_ld(Bt + c * 512, lane);
#pragma unroll
        for (int c = 0; c < 4; ++c) acc[c].v = wmma16(a, bf[c], acc[c].v);
    }

    int n = lane & 15, hi = lane >> 4;
    _Float16* Op = qt_t + (size_t)bh * NS * ND + (size_t)blockIdx.x * 4096;
#pragma unroll
    for (int c = 0; c < 4; ++c)
#pragma unroll
        for (int r = 0; r < 8; ++r) {
            int dcol = 16 * c + n;
            Op[(w * 2 + (dcol >> 5)) * 512 + fragA_idx(r + 8 * hi, dcol & 31)] =
                (_Float16)(acc[c].e[r] * 0.125f);
        }
}

// ---------------------------------------------------------------------------
// Kernel 4: fused flash attention per (b,h, 64-row q tile). Pipelined K/V.
// ---------------------------------------------------------------------------
__global__ void k_attn(const _Float16* __restrict__ qt_t, const _Float16* __restrict__ kr_t,
                       const _Float16* __restrict__ vh_t, _Float16* __restrict__ ctx_t) {
    __shared__ __align__(32) _Float16 Qt[4096];        // 8 A-frags (64x64)
    __shared__ __align__(32) _Float16 Kt[4096];        // 8 B-frags (d x key)
    __shared__ __align__(32) _Float16 Vt[4096];        // 8 B-frags (key x d)
    __shared__ __align__(32) _Float16 Pt[4 * 1024];    // per-wave 2 A-frags

    int q0 = blockIdx.x * 64, bh = blockIdx.y;
    int bb = bh >> 4, hh = bh & 15;
    int tid = threadIdx.x, lane = tid & 31, w = tid >> 5;
    int n = lane & 15, hi = lane >> 4;

    const _Float16* Qp = qt_t + (size_t)bh * NS * ND + (size_t)blockIdx.x * 4096;
    const _Float16* Kp = kr_t + (size_t)bh * NS * ND;
    const _Float16* Vp = vh_t + (size_t)bh * NS * ND;

#ifdef HAVE_TDM
    if (w == 0) {                                   // one wave drives the TDM
        tdm_copy_to_lds(Qt, Qp, 4096);
        __builtin_amdgcn_s_wait_tensorcnt(0);
    }
#else
    {
        const uint4* s4 = (const uint4*)Qp;
        uint4* d4 = (uint4*)Qt;
#pragma unroll
        for (int u = 0; u < 4; ++u) d4[tid + (u << 7)] = s4[tid + (u << 7)];
    }
#endif

    FragF accV[4];
    float m_run[8], l_run[8];
#pragma unroll
    for (int c = 0; c < 4; ++c)
#pragma unroll
        for (int e = 0; e < 8; ++e) accV[c].e[e] = 0.0f;
#pragma unroll
    for (int r = 0; r < 8; ++r) { m_run[r] = -3.0e38f; l_run[r] = 0.0f; }

    _Float16* Pw = Pt + w * 1024;

    uint4 kw[4], vw[4];
    auto load_kv = [&](int kt) {
        const uint4* sk = (const uint4*)(Kp + (size_t)(kt >> 6) * 4096);
        const uint4* sv = (const uint4*)(Vp + (size_t)(kt >> 6) * 4096);
#pragma unroll
        for (int u = 0; u < 4; ++u) {
            kw[u] = sk[tid + (u << 7)];
            vw[u] = sv[tid + (u << 7)];
        }
    };
    auto store_kv = [&]() {
        uint4* dk = (uint4*)Kt;
        uint4* dv = (uint4*)Vt;
#pragma unroll
        for (int u = 0; u < 4; ++u) {
            dk[tid + (u << 7)] = kw[u];
            dv[tid + (u << 7)] = vw[u];
        }
    };

    load_kv(0);
    for (int kt = 0; kt < NS; kt += 64) {
        __syncthreads();
        store_kv();
        __syncthreads();
        if (kt + 64 < NS) load_kv(kt + 64);   // overlaps with compute below

        // ---- scores: per wave 16 q-rows x 64 keys ----
        FragF s[4];
#pragma unroll
        for (int c = 0; c < 4; ++c)
#pragma unroll
            for (int e = 0; e < 8; ++e) s[c].e[e] = 0.0f;

        v16h qa0 = frag_ld(Qt + (w * 2 + 0) * 512, lane);
        v16h qa1 = frag_ld(Qt + (w * 2 + 1) * 512, lane);
        v16h bf[4];
#pragma unroll
        for (int c = 0; c < 4; ++c) bf[c] = frag_ld(Kt + c * 512, lane);
#pragma unroll
        for (int c = 0; c < 4; ++c) s[c].v = wmma16(qa0, bf[c], s[c].v);
#pragma unroll
        for (int c = 0; c < 4; ++c) bf[c] = frag_ld(Kt + 2048 + c * 512, lane);
#pragma unroll
        for (int c = 0; c < 4; ++c) s[c].v = wmma16(qa1, bf[c], s[c].v);

        // ---- online softmax (row = r + 8*hi, 16 cols per lane-group) ----
#pragma unroll
        for (int r = 0; r < 8; ++r) {
            float mx = s[0].e[r];
#pragma unroll
            for (int c = 1; c < 4; ++c) mx = fmaxf(mx, s[c].e[r]);
#pragma unroll
            for (int off = 1; off < 16; off <<= 1)
                mx = fmaxf(mx, __shfl_xor(mx, off, 32));
            float mn = fmaxf(m_run[r], mx);
            float scale = __expf(m_run[r] - mn);
            float rs = 0.0f;
#pragma unroll
            for (int c = 0; c < 4; ++c) {
                float p = __expf(s[c].e[r] - mn);
                s[c].e[r] = p;
                rs += p;
            }
#pragma unroll
            for (int off = 1; off < 16; off <<= 1)
                rs += __shfl_xor(rs, off, 32);
            l_run[r] = l_run[r] * scale + rs;
            m_run[r] = mn;
#pragma unroll
            for (int c = 0; c < 4; ++c) accV[c].e[r] *= scale;
        }

        // ---- scatter P into per-wave A-fragments (same-wave LDS, in-order) ----
#pragma unroll
        for (int c = 0; c < 4; ++c)
#pragma unroll
            for (int r = 0; r < 8; ++r) {
                int kcol = 16 * c + n;
                Pw[(kcol >> 5) * 512 + fragA_idx(r + 8 * hi, kcol & 31)] =
                    (_Float16)s[c].e[r];
            }

        v16h pa0 = frag_ld(Pw, lane);
        v16h pa1 = frag_ld(Pw + 512, lane);
#pragma unroll
        for (int c = 0; c < 4; ++c) bf[c] = frag_ld(Vt + c * 512, lane);
#pragma unroll
        for (int c = 0; c < 4; ++c) accV[c].v = wmma16(pa0, bf[c], accV[c].v);
#pragma unroll
        for (int c = 0; c < 4; ++c) bf[c] = frag_ld(Vt + 2048 + c * 512, lane);
#pragma unroll
        for (int c = 0; c < 4; ++c) accV[c].v = wmma16(pa1, bf[c], accV[c].v);
    }

    // ---- epilogue: ctx written pre-swizzled A-layout for k_proj ----
#pragma unroll
    for (int c = 0; c < 4; ++c)
#pragma unroll
        for (int r = 0; r < 8; ++r) {
            int row = bb * NS + q0 + 16 * w + r + 8 * hi;   // 0..4095
            int col = hh * ND + 16 * c + n;                 // 0..1023
            float o = accV[c].e[r] / l_run[r];
            ctx_t[((size_t)(row >> 6) * (DM >> 5) + (col >> 5)) * 2048
                  + ((row >> 4) & 3) * 512 + fragA_idx(row & 15, col & 31)] =
                (_Float16)o;
        }
}

// ---------------------------------------------------------------------------
// Kernel 5: out = ctx @ out_w^T + out_b   (M=4096, N=1024, K=1024). Pipelined.
// ---------------------------------------------------------------------------
__global__ void k_proj(const _Float16* __restrict__ ctx_t, const _Float16* __restrict__ wh_t,
                       const float* __restrict__ bias, float* __restrict__ out) {
    __shared__ __align__(32) _Float16 At[2048];
    __shared__ __align__(32) _Float16 Bt[2048];
    int m0 = blockIdx.x * 64, n0 = blockIdx.y * 64;
    int tid = threadIdx.x, lane = tid & 31, w = tid >> 5;

    FragF acc[4];
#pragma unroll
    for (int c = 0; c < 4; ++c)
#pragma unroll
        for (int e = 0; e < 8; ++e) acc[c].e[e] = 0.0f;

    uint4 aw[2], bw[2];
    auto load_tiles = [&](int k0) {
        const uint4* sa = (const uint4*)(ctx_t +
            ((size_t)(m0 >> 6) * (DM >> 5) + (k0 >> 5)) * 2048);
        const uint4* sb = (const uint4*)(wh_t +
            ((size_t)(k0 >> 5) * (DM >> 6) + (n0 >> 6)) * 2048);
        aw[0] = sa[tid]; aw[1] = sa[tid + 128];
        bw[0] = sb[tid]; bw[1] = sb[tid + 128];
    };
    auto store_tiles = [&]() {
        uint4* da = (uint4*)At;
        uint4* db = (uint4*)Bt;
        da[tid] = aw[0]; da[tid + 128] = aw[1];
        db[tid] = bw[0]; db[tid + 128] = bw[1];
    };

    load_tiles(0);
    for (int k0 = 0; k0 < DM; k0 += 32) {
        __syncthreads();
        store_tiles();
        __syncthreads();
        if (k0 + 32 < DM) load_tiles(k0 + 32);   // overlaps with WMMAs
        v16h a = frag_ld(At + w * 512, lane);
        v16h bf[4];
#pragma unroll
        for (int c = 0; c < 4; ++c) bf[c] = frag_ld(Bt + c * 512, lane);
#pragma unroll
        for (int c = 0; c < 4; ++c) acc[c].v = wmma16(a, bf[c], acc[c].v);
    }
    int n = lane & 15, hi = lane >> 4;
#pragma unroll
    for (int c = 0; c < 4; ++c)
#pragma unroll
        for (int r = 0; r < 8; ++r) {
            int row = m0 + 16 * w + r + 8 * hi;
            int col = n0 + 16 * c + n;
            out[(size_t)row * DM + col] = acc[c].e[r] + bias[col];
        }
}

// ---------------------------------------------------------------------------
extern "C" void kernel_launch(void* const* d_in, const int* in_sizes, int n_in,
                              void* d_out, int out_size, void* d_ws, size_t ws_size,
                              hipStream_t stream) {
    const float* q     = (const float*)d_in[0];
    const float* k     = (const float*)d_in[1];
    const float* v     = (const float*)d_in[2];
    const float* mask  = (const float*)d_in[3];
    const float* out_w = (const float*)d_in[4];
    const float* out_b = (const float*)d_in[5];
    float* out = (float*)d_out;

    const size_t NELT = (size_t)NB * NH * NS * ND;   // 4M
    _Float16* ws    = (_Float16*)d_ws;
    _Float16* qr_t  = ws;                 // 4M halves (tiled)
    _Float16* kr_t  = qr_t  + NELT;       // 4M
    _Float16* vh_t  = kr_t  + NELT;       // 4M
    _Float16* qt_t  = vh_t  + NELT;       // 4M
    _Float16* ctx_t = qt_t  + NELT;       // 4M
    _Float16* wh_t  = ctx_t + NELT;       // 1M
    // total workspace: 21M halves = 42 MB

    k_rope<<<(NELT + 255) / 256, 256, 0, stream>>>(q, k, v, qr_t, kr_t, vh_t);
    k_cvtw<<<(DM * DM + 255) / 256, 256, 0, stream>>>(out_w, wh_t, DM * DM);
    k_qt  <<<dim3(NS / 64, NB * NH), 128, 0, stream>>>(mask, qr_t, qt_t);
    k_attn<<<dim3(NS / 64, NB * NH), 128, 0, stream>>>(qt_t, kr_t, vh_t, ctx_t);
    k_proj<<<dim3(NB * NS / 64, DM / 64), 128, 0, stream>>>(ctx_t, wh_t, out_b, out);
}
